// MultiHeadAttention_4861902979353
// MI455X (gfx1250) — matrix operand order repaired and checked by reading, not compiled, not run on hardware
//
#include <hip/hip_runtime.h>
#include <math.h>
#include <stdint.h>

// Problem constants (from the reference).
#define BB   2
#define SS   2048
#define DD   1024
#define HH   16
#define DHD  64
#define SPP  1024
#define INV_SCALE 0.125f   // 1/sqrt(64)

typedef __attribute__((ext_vector_type(16))) __bf16 v16bf;
typedef __attribute__((ext_vector_type(8)))  float  v8f;

union FragU { v16bf v; uint4 q[2]; };

#define WMMA_BF16(a, b, c) \
  __builtin_amdgcn_wmma_f32_16x16x32_bf16(false, (a), false, (b), (short)0, (c), false, false)

// ---------------------------------------------------------------------------
// Fragment loaders (wave32); contiguous k order k = 16*(l>>4) + s, same in A and B.
// A (16x32 MxK, bf16): lane l<16 -> row l, K = kb..kb+15, kb=0
//                      lane l>=16 -> row l-16, kb=16. Two 16B loads.
// B (32x16 KxN, bf16): lane l -> col l&15, K = (l>>4)*16 .. +15. One 32B run.
// ---------------------------------------------------------------------------
static __device__ __forceinline__
v16bf load_afrag(const __bf16* A, int ldk, int row0, int k0, int lane) {
  const __bf16* p = A + (size_t)(row0 + (lane & 15)) * ldk + (k0 + ((lane >> 4) << 4));
  FragU f;
  f.q[0] = *(const uint4*)p;          // K kb..kb+7
  f.q[1] = *(const uint4*)(p + 8);    // K kb+8..kb+15
  return f.v;
}

static __device__ __forceinline__
v16bf load_bfrag(const __bf16* Wm, int ldk, int n0, int k0, int lane) {
  const __bf16* p = Wm + (size_t)(n0 + (lane & 15)) * ldk + (k0 + ((lane >> 4) << 4));
  FragU f;
  f.q[0] = *(const uint4*)p;          // K kb..kb+7
  f.q[1] = *(const uint4*)(p + 8);    // K kb+8..kb+15
  return f.v;
}

// B fragment from an LDS-staged tile laid out [64 rows][32 k] bf16, row-major.
static __device__ __forceinline__
v16bf lds_bfrag(const __bf16* buf, int t, int lane) {
  const __bf16* p = buf + (t * 16 + (lane & 15)) * 32 + ((lane >> 4) << 4);
  FragU f;
  f.q[0] = *(const uint4*)p;
  f.q[1] = *(const uint4*)(p + 8);
  return f.v;
}

// ---------------------------------------------------------------------------
// Async staging of the block-shared B tile (64 rows x 32 k, bf16 = 4KB) into
// LDS via CDNA5 GLOBAL_LOAD_ASYNC_TO_LDS_B128 (tracked on ASYNCcnt).
// 128 threads x 2 chunks x 16B = 4KB.  Chunk c: row = c>>2, kc = (c&3)*8.
// ---------------------------------------------------------------------------
static __device__ __forceinline__
void copy_btile_async(__bf16* buf, const __bf16* W, int ldk, int n0, int k0, int tid) {
#pragma unroll
  for (int j = 0; j < 2; ++j) {
    const int c   = tid + 128 * j;
    const int row = c >> 2;
    const int kc  = (c & 3) << 3;
    const unsigned long long ga =
        (unsigned long long)(const void*)(W + (size_t)(n0 + row) * ldk + (k0 + kc));
    const unsigned la = (unsigned)(uintptr_t)(buf + row * 32 + kc);
    asm volatile("global_load_async_to_lds_b128 %0, %1, off"
                 :: "v"(la), "v"(ga) : "memory");
  }
}
#define WAIT_ASYNC_2() asm volatile("s_wait_asynccnt 0x2" ::: "memory")
#define WAIT_ASYNC_0() asm volatile("s_wait_asynccnt 0x0" ::: "memory")

// ---------------------------------------------------------------------------
// f32 -> bf16 bulk conversion (float4 in, 4x bf16 out).
// ---------------------------------------------------------------------------
__global__ void k_cvt(const float* __restrict__ src, __bf16* __restrict__ dst, int n4) {
  int i = blockIdx.x * blockDim.x + threadIdx.x;
  int stride = gridDim.x * blockDim.x;
  for (; i < n4; i += stride) {
    float4 f = ((const float4*)src)[i];
    union { __bf16 h[4]; uint2 u; } o;
    o.h[0] = (__bf16)f.x; o.h[1] = (__bf16)f.y;
    o.h[2] = (__bf16)f.z; o.h[3] = (__bf16)f.w;
    *(uint2*)(dst + 4 * (size_t)i) = o.u;
  }
}

// ---------------------------------------------------------------------------
// Generic GEMM: C[M,N] = A[M,K] @ W[N,K]^T (+bias[n]), 4 waves/block,
// each wave owns a 16x64 strip; the block-shared W tile is double-buffered
// in LDS via async copies. Epilogue modes:
//   0: Q store  -> bf16 [B,H,S,DH], val = (acc+b) * mask[row] * scale
//   1: K/V store-> bf16 [B,H,DH,S] (transposed per head), val = (acc+b)*mask
//   2: f32 row-major store (output projection)
// ---------------------------------------------------------------------------
__global__ __launch_bounds__(128)
void k_gemm(const __bf16* __restrict__ A, const __bf16* __restrict__ W,
            const float* __restrict__ bias, const int* __restrict__ mask,
            float* __restrict__ outF, __bf16* __restrict__ outB,
            int M, int N, int K, int mode, float scale) {
  __shared__ __align__(16) __bf16 Bt[2][64 * 32];
  const int tid = threadIdx.x;
  const int lane = tid & 31, wave = tid >> 5;
  const int row0 = blockIdx.x * 64 + wave * 16;
  const int col0 = blockIdx.y * 64;

  copy_btile_async(Bt[0], W, K, col0, 0, tid);

  v8f acc[4] = {};
  for (int k0 = 0; k0 < K; k0 += 32) {
    const int cur = (k0 >> 5) & 1;
    const bool more = (k0 + 32 < K);
    if (more) {
      copy_btile_async(Bt[cur ^ 1], W, K, col0, k0 + 32, tid);
      WAIT_ASYNC_2();                // this wave's copies for `cur` landed
    } else {
      WAIT_ASYNC_0();
    }
    __syncthreads();                 // all waves' portions of Bt[cur] visible
    if (k0 + 64 < K)                 // near-cache prefetch of the A stream
      __builtin_prefetch(A + (size_t)(row0 + (lane & 15)) * K + k0 + 64, 0, 3);
    v16bf a = load_afrag(A, K, row0, k0, lane);
#pragma unroll
    for (int t = 0; t < 4; ++t)
      acc[t] = WMMA_BF16(a, lds_bfrag(Bt[cur], t, lane), acc[t]);
    __syncthreads();                 // reads done before Bt[cur] is recycled
  }

  const int half = lane >> 4, nl = lane & 15;
#pragma unroll
  for (int t = 0; t < 4; ++t) {
    const int n = col0 + t * 16 + nl;
    const float bv = bias ? bias[n] : 0.0f;
#pragma unroll
    for (int r = 0; r < 8; ++r) {
      const int m = row0 + r + 8 * half;        // C layout: lane half selects M+8
      float v = acc[t][r] + bv;
      if (mode <= 1) {
        v *= scale * (float)(mask[m] != 0);     // mask is per (b,s) row
        const int b = m / SS, s = m % SS, h = n / DHD, d = n % DHD;
        const size_t idx = (mode == 0)
            ? ((((size_t)b * HH + h) * SS + s) * DHD + d)    // [B,H,S,DH]
            : ((((size_t)b * HH + h) * DHD + d) * SS + s);   // [B,H,DH,S]
        outB[idx] = (__bf16)v;
      } else {
        outF[(size_t)m * N + n] = v;
      }
    }
  }
}

// ---------------------------------------------------------------------------
// Linformer projection: per (b,h): P[p,d] = gelu( sum_s WP[p,s]*XT[d,s] + bias[p] )
//   M=SP, N=DH, K=S. B tile (all 64 rows of XT-head) async-staged in LDS.
//   transOut=0 -> [BH,SP,DH] (Kp); 1 -> [BH,DH,SP] (VpT)
// ---------------------------------------------------------------------------
__global__ __launch_bounds__(128)
void k_linproj(const __bf16* __restrict__ WP, const __bf16* __restrict__ XT,
               const float* __restrict__ biasRow, __bf16* __restrict__ out,
               int transOut) {
  __shared__ __align__(16) __bf16 Bt[2][64 * 32];
  const int tid = threadIdx.x;
  const int lane = tid & 31, wave = tid >> 5;
  const int p0 = blockIdx.x * 64 + wave * 16;
  const int bh = blockIdx.y;
  const __bf16* Xh = XT + (size_t)bh * DHD * SS;

  copy_btile_async(Bt[0], Xh, SS, 0, 0, tid);

  v8f acc[4] = {};
  for (int k0 = 0; k0 < SS; k0 += 32) {
    const int cur = (k0 >> 5) & 1;
    const bool more = (k0 + 32 < SS);
    if (more) {
      copy_btile_async(Bt[cur ^ 1], Xh, SS, 0, k0 + 32, tid);
      WAIT_ASYNC_2();
    } else {
      WAIT_ASYNC_0();
    }
    __syncthreads();
    if (k0 + 64 < SS)
      __builtin_prefetch(WP + (size_t)(p0 + (lane & 15)) * SS + k0 + 64, 0, 3);
    v16bf a = load_afrag(WP, SS, p0, k0, lane);
#pragma unroll
    for (int t = 0; t < 4; ++t)
      acc[t] = WMMA_BF16(a, lds_bfrag(Bt[cur], t, lane), acc[t]);
    __syncthreads();
  }

  const int half = lane >> 4, nl = lane & 15;
#pragma unroll
  for (int t = 0; t < 4; ++t) {
    const int d = t * 16 + nl;
#pragma unroll
    for (int r = 0; r < 8; ++r) {
      const int p = p0 + r + 8 * half;
      float v = acc[t][r] + biasRow[p];                       // bias is per ROW p
      v = 0.5f * v * (1.0f + erff(v * 0.70710678118654752f)); // exact GELU
      const size_t idx = transOut
          ? ((size_t)bh * DHD + d) * SPP + p
          : ((size_t)bh * SPP + p) * DHD + d;
      out[idx] = (__bf16)v;
    }
  }
}

// ---------------------------------------------------------------------------
// Flash-style attention per (b,h): each wave owns 16 query rows, online
// softmax over SP in p-tiles of 32. Q is pre-scaled by mask/sqrt(dh).
//   Q  [BH,S,DH] bf16,  Kp [BH,SP,DH] bf16,  VpT [BH,DH,SP] bf16
//   X output: bf16 [B,S,D] (row-major, feeds the output projection GEMM)
// ---------------------------------------------------------------------------
__global__ __launch_bounds__(128)
void k_attn(const __bf16* __restrict__ Q, const __bf16* __restrict__ Kp,
            const __bf16* __restrict__ VpT, __bf16* __restrict__ X) {
  __shared__ __align__(16) __bf16 lds[4][16 * 32];  // per-wave P tile (bf16)
  const int lane = threadIdx.x & 31, wave = threadIdx.x >> 5;
  const int q0 = blockIdx.x * 64 + wave * 16;
  const int bh = blockIdx.y;
  const int half = lane >> 4, nl = lane & 15;

  const __bf16* Qh = Q   + (size_t)bh * SS  * DHD;
  const __bf16* Kh = Kp  + (size_t)bh * SPP * DHD;
  const __bf16* Vh = VpT + (size_t)bh * DHD * SPP;
  __bf16* myP = lds[wave];

  // Q fragments for this wave's 16 rows (d = 0..31 and 32..63), loaded once.
  const v16bf aq0 = load_afrag(Qh, DHD, q0, 0,  lane);
  const v16bf aq1 = load_afrag(Qh, DHD, q0, 32, lane);

  v8f o[4] = {};
  float mrun[8], lrun[8];
#pragma unroll
  for (int r = 0; r < 8; ++r) { mrun[r] = -3.0e38f; lrun[r] = 0.0f; }

  for (int p0 = 0; p0 < SPP; p0 += 32) {
    // scores: two 16x16 tiles, K = d (64) in two WMMA steps each
    v8f s0 = {}, s1 = {};
    s0 = WMMA_BF16(aq0, load_bfrag(Kh, DHD, p0,      0,  lane), s0);
    s0 = WMMA_BF16(aq1, load_bfrag(Kh, DHD, p0,      32, lane), s0);
    s1 = WMMA_BF16(aq0, load_bfrag(Kh, DHD, p0 + 16, 0,  lane), s1);
    s1 = WMMA_BF16(aq1, load_bfrag(Kh, DHD, p0 + 16, 32, lane), s1);

    // online softmax update per row (row = r + 8*half, spread over 16 lanes)
#pragma unroll
    for (int r = 0; r < 8; ++r) {
      float rmax = fmaxf(s0[r], s1[r]);
#pragma unroll
      for (int xm = 8; xm >= 1; xm >>= 1) rmax = fmaxf(rmax, __shfl_xor(rmax, xm, 32));
      const float mn   = fmaxf(mrun[r], rmax);
      const float corr = __expf(mrun[r] - mn);
      const float e0 = __expf(s0[r] - mn);
      const float e1 = __expf(s1[r] - mn);
      float rsum = e0 + e1;
#pragma unroll
      for (int xm = 8; xm >= 1; xm >>= 1) rsum += __shfl_xor(rsum, xm, 32);
      lrun[r] = lrun[r] * corr + rsum;
      mrun[r] = mn;
      o[0][r] *= corr; o[1][r] *= corr; o[2][r] *= corr; o[3][r] *= corr;
      // C-layout -> LDS row-major P tile (16 x 32 bf16)
      myP[(r + 8 * half) * 32 + nl]      = (__bf16)e0;
      myP[(r + 8 * half) * 32 + nl + 16] = (__bf16)e1;
    }
    __syncthreads();                         // order DS stores before DS loads
    const v16bf ap = load_afrag(myP, 32, 0, 0, lane);   // P as A-operand
    __syncthreads();

    // O += P @ Vp  (K = 32 p-values, 4 n-tiles of d)
#pragma unroll
    for (int t = 0; t < 4; ++t) {
      v16bf bv = load_bfrag(Vh, SPP, t * 16, p0, lane);
      o[t] = WMMA_BF16(ap, bv, o[t]);
    }
  }

  // normalize and store to [B,S,D] bf16 (feeds output projection)
  const int b = bh / HH, h = bh % HH;
#pragma unroll
  for (int t = 0; t < 4; ++t) {
    const int d = t * 16 + nl;
#pragma unroll
    for (int r = 0; r < 8; ++r) {
      const int s = q0 + r + 8 * half;
      const float v = o[t][r] / lrun[r];
      X[((size_t)b * SS + s) * DD + h * DHD + d] = (__bf16)v;
    }
  }
}

// ---------------------------------------------------------------------------
extern "C" void kernel_launch(void* const* d_in, const int* in_sizes, int n_in,
                              void* d_out, int out_size, void* d_ws, size_t ws_size,
                              hipStream_t stream) {
  (void)in_sizes; (void)n_in; (void)out_size; (void)ws_size;

  const float* query = (const float*)d_in[0];
  const float* keyi  = (const float*)d_in[1];
  const float* value = (const float*)d_in[2];
  const int*   mask  = (const int*)  d_in[3];    // [B,1,1,S] -> flat B*S
  const float* Wq = (const float*)d_in[4];  const float* bq = (const float*)d_in[5];
  const float* Wk = (const float*)d_in[6];  const float* bk = (const float*)d_in[7];
  const float* Wv = (const float*)d_in[8];  const float* bv = (const float*)d_in[9];
  const float* Wo = (const float*)d_in[10]; const float* bo = (const float*)d_in[11];
  const float* WkP = (const float*)d_in[12]; const float* bkP = (const float*)d_in[13];
  const float* WvP = (const float*)d_in[14]; const float* bvP = (const float*)d_in[15];

  const size_t EL_IN = (size_t)BB * SS * DD;       // 4.19M
  const size_t EL_W  = (size_t)DD * DD;            // 1.05M
  const size_t EL_WP = (size_t)SPP * SS;           // 2.10M
  const size_t EL_P  = (size_t)BB * HH * SPP * DHD;// 2.10M

  __bf16* w = (__bf16*)d_ws;
  __bf16* qc  = w; w += EL_IN;   // bf16 copies of inputs
  __bf16* kc  = w; w += EL_IN;
  __bf16* vc  = w; w += EL_IN;
  __bf16* wqb = w; w += EL_W;    // bf16 weights
  __bf16* wkb = w; w += EL_W;
  __bf16* wvb = w; w += EL_W;
  __bf16* wob = w; w += EL_W;
  __bf16* wkp = w; w += EL_WP;
  __bf16* wvp = w; w += EL_WP;
  __bf16* Qb  = w; w += EL_IN;   // [B,H,S,DH], pre-scaled by mask/sqrt(dh)
  __bf16* KT  = w; w += EL_IN;   // [B,H,DH,S]
  __bf16* VT  = w; w += EL_IN;   // [B,H,DH,S]
  __bf16* Kpb = w; w += EL_P;    // [BH,SP,DH]
  __bf16* VpT = w; w += EL_P;    // [BH,DH,SP]
  __bf16* xb  = w; w += EL_IN;   // [B,S,D] attention output, bf16

  // 1) fp32 -> bf16 conversions
  const float* srcs[9] = { query, keyi, value, Wq, Wk, Wv, Wo, WkP, WvP };
  __bf16* dsts[9]      = { qc, kc, vc, wqb, wkb, wvb, wob, wkp, wvp };
  const size_t cnts[9] = { EL_IN, EL_IN, EL_IN, EL_W, EL_W, EL_W, EL_W, EL_WP, EL_WP };
  for (int i = 0; i < 9; ++i)
    k_cvt<<<1024, 256, 0, stream>>>(srcs[i], dsts[i], (int)(cnts[i] / 4));

  const dim3 blk(128);
  const dim3 gQKV((BB * SS) / 64, DD / 64);        // 64 x 16

  // 2) QKV projections (mask fused; Q also pre-scaled by 1/sqrt(dh))
  k_gemm<<<gQKV, blk, 0, stream>>>(qc, wqb, bq, mask, nullptr, Qb,
                                   BB * SS, DD, DD, 0, INV_SCALE);
  k_gemm<<<gQKV, blk, 0, stream>>>(kc, wkb, bk, mask, nullptr, KT,
                                   BB * SS, DD, DD, 1, 1.0f);
  k_gemm<<<gQKV, blk, 0, stream>>>(vc, wvb, bv, mask, nullptr, VT,
                                   BB * SS, DD, DD, 1, 1.0f);

  // 3) Linformer seq projections with exact-GELU epilogue
  const dim3 gLP(SPP / 64, BB * HH);               // 16 x 32
  k_linproj<<<gLP, blk, 0, stream>>>(wkp, KT, bkP, Kpb, 0);
  k_linproj<<<gLP, blk, 0, stream>>>(wvp, VT, bvP, VpT, 1);

  // 4) fused attention (score -> online softmax -> PV)
  const dim3 gAT(SS / 64, BB * HH);                // 32 x 32
  k_attn<<<gAT, blk, 0, stream>>>(Qb, Kpb, VpT, xb);

  // 5) output projection, f32 result
  k_gemm<<<gQKV, blk, 0, stream>>>(xb, wob, bo, nullptr, (float*)d_out, nullptr,
                                   BB * SS, DD, DD, 2, 1.0f);
}
